// CNN_ternary_32487132627221
// MI455X (gfx1250) — 8-bit matrix operand order repaired and checked by reading, compiled, not run on hardware
//
#include <hip/hip_runtime.h>
#include <hip/hip_bf16.h>

typedef __attribute__((ext_vector_type(8))) int v8i;

#define TDELTA 0.1f
#define BN_EPS 1e-5f

#define OFF_W2P   0
#define OFF_W3P   6144
#define OFF_W4P   30720
#define OFF_WFP   129024
#define OFF_W1T   161792
#define OFF_SQ    162944
#define OFF_TBF   165760
#define PACK_TOTAL 162448

__device__ __forceinline__ float ternf(float t) {
    return (fabsf(t) < TDELTA) ? 0.0f : (t > 0.0f ? 1.0f : -1.0f);
}
__device__ __forceinline__ signed char tern8(float t) {
    return (fabsf(t) < TDELTA) ? (signed char)0 : (t > 0.0f ? (signed char)1 : (signed char)-1);
}

__global__ void tnn_pack(const float* __restrict__ w1, const float* __restrict__ b1,
                         const float* __restrict__ w2, const float* __restrict__ b2,
                         const float* __restrict__ w3, const float* __restrict__ b3,
                         const float* __restrict__ w4, const float* __restrict__ b4,
                         const float* __restrict__ wf, const float* __restrict__ bf,
                         const float* __restrict__ g1, const float* __restrict__ be1,
                         const float* __restrict__ m1, const float* __restrict__ v1,
                         const float* __restrict__ g2, const float* __restrict__ be2,
                         const float* __restrict__ m2, const float* __restrict__ v2,
                         const float* __restrict__ g3, const float* __restrict__ be3,
                         const float* __restrict__ m3, const float* __restrict__ v3,
                         const float* __restrict__ g4, const float* __restrict__ be4,
                         const float* __restrict__ m4, const float* __restrict__ v4,
                         unsigned char* __restrict__ ws) {
    signed char* w2p = (signed char*)(ws + OFF_W2P);
    signed char* w3p = (signed char*)(ws + OFF_W3P);
    signed char* w4p = (signed char*)(ws + OFF_W4P);
    signed char* wfp = (signed char*)(ws + OFF_WFP);
    float* w1t = (float*)(ws + OFF_W1T);
    float* sq  = (float*)(ws + OFF_SQ);
    float* tbf = (float*)(ws + OFF_TBF);

    for (int idx = blockIdx.x * blockDim.x + threadIdx.x; idx < PACK_TOTAL;
         idx += gridDim.x * blockDim.x) {
        if (idx < 6144) {
            int oc = idx / 96, k = idx % 96;
            int kx = k >> 5, ic = k & 31;
            w2p[idx] = tern8(w2[oc * 96 + ic * 3 + kx]);
        } else if (idx < 30720) {
            int j = idx - 6144;
            int oc = j / 192, k = j % 192;
            int kx = k >> 6, ic = k & 63;
            w3p[j] = tern8(w3[oc * 192 + ic * 3 + kx]);
        } else if (idx < 129024) {
            int j = idx - 30720;
            int oc = j / 768, k = j % 768;
            int kh = k >> 7, ic = k & 127;
            w4p[j] = tern8(w4[oc * 768 + ic * 6 + kh]);
        } else if (idx < 161792) {
            int j = idx - 129024;
            int row = j >> 11, k = j & 2047;
            wfp[j] = (row < 10) ? tern8(wf[row * 2048 + k]) : (signed char)0;
        } else if (idx < 162080) {
            int j = idx - 161792;
            w1t[j] = ternf(w1[j]);
        } else if (idx < 162432) {
            int j = idx - 162080;
            float g, be, m, v, tb;
            if (j < 32)       { int c = j;        g = g1[c]; be = be1[c]; m = m1[c]; v = v1[c]; tb = ternf(b1[c]); }
            else if (j < 96)  { int c = j - 32;   g = g2[c]; be = be2[c]; m = m2[c]; v = v2[c]; tb = ternf(b2[c]); }
            else if (j < 224) { int c = j - 96;   g = g3[c]; be = be3[c]; m = m3[c]; v = v3[c]; tb = ternf(b3[c]); }
            else              { int c = j - 224;  g = g4[c]; be = be4[c]; m = m4[c]; v = v4[c]; tb = ternf(b4[c]); }
            float s = g * rsqrtf(v + BN_EPS);
            sq[j * 2 + 0] = s;
            sq[j * 2 + 1] = tb * s + (be - m * s);
        } else {
            int j = idx - 162432;
            tbf[j] = (j < 10) ? ternf(bf[j]) : 0.0f;
        }
    }
}

__device__ __forceinline__ v8i make_a(const signed char* __restrict__ rowptr,
                                      int kc, int ktot, bool hi) {
    v8i a;
#pragma unroll
    for (int v = 0; v < 8; ++v) {
        int kb = kc + ((v >> 1) << 4) + ((v & 1) << 2) + (hi ? 8 : 0);
        a[v] = (kb < ktot) ? *(const int*)(rowptr + kb) : 0;
    }
    return a;
}
__device__ __forceinline__ int kbase_b(int v, bool hi) {
    return ((v >> 1) << 4) + ((v & 1) << 2) + (hi ? 8 : 0);
}

__global__ __launch_bounds__(256) void tnn_fused(const float* __restrict__ x,
                                                 const unsigned char* __restrict__ ws,
                                                 float* __restrict__ out) {
    const signed char* w2p = (const signed char*)(ws + OFF_W2P);
    const signed char* w3p = (const signed char*)(ws + OFF_W3P);
    const signed char* w4p = (const signed char*)(ws + OFF_W4P);
    const signed char* wfp = (const signed char*)(ws + OFF_WFP);
    const float* w1t = (const float*)(ws + OFF_W1T);
    const float* sq1 = (const float*)(ws + OFF_SQ);
    const float* sq2 = sq1 + 32 * 2;
    const float* sq3 = sq1 + 96 * 2;
    const float* sq4 = sq1 + 224 * 2;
    const float* tbf = (const float*)(ws + OFF_TBF);

    __shared__ int sA[6144];
    __shared__ int sB[6528];
    __shared__ int sH4[1024];
    char* h1 = (char*)sA;
    char* h3 = (char*)sA;
    char* h2 = (char*)sB;
    char* h4 = (char*)sH4;

    const int tid  = threadIdx.x;
    const int lane = tid & 31;
    const int wave = tid >> 5;
    const bool hi  = lane >= 16;
    const int e0   = blockIdx.x * 2;

    for (int i = tid; i < 6144; i += 256) sA[i] = 0;
    for (int i = tid; i < 6528; i += 256) sB[i] = 0;
    __syncthreads();

    for (int i = tid; i < 12288; i += 256) {
        int e = i / 6144, rem = i - e * 6144;
        int h = rem >> 10, wo = (rem >> 5) & 31, c = rem & 31;
        const float* xr = x + ((e0 + e) * 6 + h) * 128;
        const float* wr = w1t + c * 9;
        float z[2];
#pragma unroll
        for (int p = 0; p < 2; ++p) {
            int cb = 4 * wo + 2 * p - 4;
            float acc = 0.0f;
#pragma unroll
            for (int k = 0; k < 9; ++k) {
                int col = cb + k;
                float xv = (col >= 0 && col < 128) ? xr[col] : 0.0f;
                acc = fmaf(wr[k], xv, acc);
            }
            z[p] = fmaf(acc, sq1[c * 2], sq1[c * 2 + 1]);
        }
        h1[((e * 6 + h) * 34 + wo + 1) * 32 + c] = tern8(fmaxf(z[0], z[1]));
    }
    __syncthreads();

    for (int t = wave; t < 96; t += 8) {
        int mt = t & 3, nt = t >> 2;
        int n = nt * 16 + (lane & 15);
        int e = n / 192, rem = n % 192;
        int h = rem >> 5, wc = rem & 31, wp = wc + 1;
        int rowBase = (e * 6 + h) * 34;
        const signed char* arow = w2p + (lane & 15) * 96 + mt * 16 * 96;
        v8i acc = {0, 0, 0, 0, 0, 0, 0, 0};
#pragma unroll
        for (int kc = 0; kc < 96; kc += 64) {
            v8i a = make_a(arow, kc, 96, hi);
            v8i b;
#pragma unroll
            for (int v = 0; v < 8; ++v) {
                int kb = kc + kbase_b(v, hi);
                int val = 0;
                if (kb < 96) {
                    int kx = kb >> 5, ic = kb & 31;
                    val = *(const int*)(h1 + ((rowBase + wp - 1 + kx) << 5) + ic);
                }
                b[v] = val;
            }
            acc = __builtin_amdgcn_wmma_i32_16x16x64_iu8(true, a, true, b, acc, false, false);
        }
#pragma unroll
        for (int r = 0; r < 8; ++r) {
            int oc = mt * 16 + r + (hi ? 8 : 0);
            float z = fmaf((float)acc[r], sq2[oc * 2], sq2[oc * 2 + 1]);
            h2[(rowBase + wp) * 64 + oc] = tern8(z);
        }
    }
    __syncthreads();

    for (int t = wave; t < 192; t += 8) {
        int mt = t & 7, nt = t >> 3;
        int n = nt * 16 + (lane & 15);
        int e = n / 192, rem = n % 192;
        int h = rem >> 5, wc = rem & 31, wp = wc + 1;
        int rowBase = (e * 6 + h) * 34;
        const signed char* arow = w3p + (lane & 15) * 192 + mt * 16 * 192;
        v8i acc = {0, 0, 0, 0, 0, 0, 0, 0};
#pragma unroll
        for (int kc = 0; kc < 192; kc += 64) {
            v8i a = make_a(arow, kc, 192, hi);
            v8i b;
#pragma unroll
            for (int v = 0; v < 8; ++v) {
                int kb = kc + kbase_b(v, hi);
                int kx = kb >> 6, ic = kb & 63;
                b[v] = *(const int*)(h2 + ((rowBase + wp - 1 + kx) << 6) + ic);
            }
            acc = __builtin_amdgcn_wmma_i32_16x16x64_iu8(true, a, true, b, acc, false, false);
        }
#pragma unroll
        for (int r = 0; r < 8; ++r) {
            int oc = mt * 16 + r + (hi ? 8 : 0);
            float z = fmaf((float)acc[r], sq3[oc * 2], sq3[oc * 2 + 1]);
            float zp = fmaxf(z, __shfl_xor(z, 1, 32));
            if ((lane & 1) == 0) {
                int wo = wc >> 1;
                h3[((e * 6 + h) * 16 + wo) * 128 + oc] = tern8(zp);
            }
        }
    }
    __syncthreads();

    for (int t = wave; t < 16; t += 8) {
        int mt = t & 7, nt = t >> 3;
        int n = nt * 16 + (lane & 15);
        int e = n >> 4, w = n & 15;
        const signed char* arow = w4p + (lane & 15) * 768 + mt * 16 * 768;
        v8i acc = {0, 0, 0, 0, 0, 0, 0, 0};
#pragma unroll
        for (int kc = 0; kc < 768; kc += 64) {
            v8i a = make_a(arow, kc, 768, hi);
            v8i b;
#pragma unroll
            for (int v = 0; v < 8; ++v) {
                int kb = kc + kbase_b(v, hi);
                int kh = kb >> 7, ic = kb & 127;
                b[v] = *(const int*)(h3 + (((e * 6 + kh) * 16 + w) << 7) + ic);
            }
            acc = __builtin_amdgcn_wmma_i32_16x16x64_iu8(true, a, true, b, acc, false, false);
        }
#pragma unroll
        for (int r = 0; r < 8; ++r) {
            int oc = mt * 16 + r + (hi ? 8 : 0);
            float z = fmaf((float)acc[r], sq4[oc * 2], sq4[oc * 2 + 1]);
            h4[e * 2048 + oc * 16 + w] = tern8(z);
        }
    }
    __syncthreads();

    if (wave == 0) {
        int n = lane & 15;
        const signed char* arow = wfp + (lane & 15) * 2048;
        v8i acc = {0, 0, 0, 0, 0, 0, 0, 0};
        for (int kc = 0; kc < 2048; kc += 64) {
            v8i a = make_a(arow, kc, 2048, hi);
            v8i b;
#pragma unroll
            for (int v = 0; v < 8; ++v) {
                int kb = kc + kbase_b(v, hi);
                b[v] = (n < 2) ? *(const int*)(h4 + n * 2048 + kb) : 0;
            }
            acc = __builtin_amdgcn_wmma_i32_16x16x64_iu8(true, a, true, b, acc, false, false);
        }
#pragma unroll
        for (int r = 0; r < 8; ++r) {
            int m = r + (hi ? 8 : 0);
            if (n < 2 && m < 10)
                out[(e0 + n) * 10 + m] = (float)acc[r] + tbf[m];
        }
    }
}

extern "C" void kernel_launch(void* const* d_in, const int* in_sizes, int n_in,
                              void* d_out, int out_size, void* d_ws, size_t ws_size,
                              hipStream_t stream) {
    const float* x  = (const float*)d_in[0];
    const float* w1 = (const float*)d_in[1];
    const float* b1 = (const float*)d_in[2];
    const float* w2 = (const float*)d_in[3];
    const float* b2 = (const float*)d_in[4];
    const float* w3 = (const float*)d_in[5];
    const float* b3 = (const float*)d_in[6];
    const float* w4 = (const float*)d_in[7];
    const float* b4 = (const float*)d_in[8];
    const float* wf = (const float*)d_in[9];
    const float* bf = (const float*)d_in[10];
    const float* g1 = (const float*)d_in[11];
    const float* be1 = (const float*)d_in[12];
    const float* m1 = (const float*)d_in[13];
    const float* v1 = (const float*)d_in[14];
    const float* g2 = (const float*)d_in[15];
    const float* be2 = (const float*)d_in[16];
    const float* m2 = (const float*)d_in[17];
    const float* v2 = (const float*)d_in[18];
    const float* g3 = (const float*)d_in[19];
    const float* be3 = (const float*)d_in[20];
    const float* m3 = (const float*)d_in[21];
    const float* v3 = (const float*)d_in[22];
    const float* g4 = (const float*)d_in[23];
    const float* be4 = (const float*)d_in[24];
    const float* m4 = (const float*)d_in[25];
    const float* v4 = (const float*)d_in[26];

    unsigned char* ws = (unsigned char*)d_ws;
    float* outp = (float*)d_out;

    tnn_pack<<<(PACK_TOTAL + 255) / 256, 256, 0, stream>>>(
        w1, b1, w2, b2, w3, b3, w4, b4, wf, bf,
        g1, be1, m1, v1, g2, be2, m2, v2,
        g3, be3, m3, v3, g4, be4, m4, v4, ws);

    tnn_fused<<<2048, 256, 0, stream>>>(x, ws, outp);
}
